// Self_27762668601709
// MI455X (gfx1250) — compile-verified, matrix operand order checked by reading, not run on hardware
//
#include <hip/hip_runtime.h>
#include <math.h>

typedef __attribute__((ext_vector_type(16))) __bf16 v16bf;
typedef __attribute__((ext_vector_type(8)))  __bf16 v8bf;
typedef __attribute__((ext_vector_type(2)))  __bf16 v2bf;
typedef __attribute__((ext_vector_type(8)))  float  v8f;
typedef __attribute__((ext_vector_type(4)))  unsigned int v4u;
typedef __attribute__((ext_vector_type(8)))  int v8i;
typedef __attribute__((ext_vector_type(4)))  int v4i;

#define WMMA_BF16(A, B, C) \
  __builtin_amdgcn_wmma_f32_16x16x32_bf16(false, (A), false, (B), (short)0, (C), false, false)

static __device__ __forceinline__ v16bf cat8(v8bf a, v8bf b) {
  v16bf r;
#pragma unroll
  for (int i = 0; i < 8; ++i) { r[i] = a[i]; r[i + 8] = b[i]; }
  return r;
}
static __device__ __forceinline__ v16bf lds2(const __bf16* p0, const __bf16* p1) {
  return cat8(*(const v8bf*)p0, *(const v8bf*)p1);
}

// ---- Tensor Data Mover: 2-D bf16 tile (64 rows x 64 elems) -> LDS with 16B row pad (stride 72) ----
// D# per CDNA5 ISA ch.8: group0 = {ctrl, lds_addr, global_addr, type=2}
//                        group1 = {flags/pad, dims, tile dims, dim0 stride}
// This toolchain exposes the 6-arg builtin: (g0, g1, g2, g3, g4, cpol).
static __device__ __forceinline__ void tdm_load_2d(unsigned lds_off,
                                                   unsigned long long gaddr,
                                                   unsigned stride_elems) {
  v4u g0;
  g0[0] = 1u;                                              // count=1 user descriptor
  g0[1] = lds_off;                                         // lds_addr (bytes)
  g0[2] = (unsigned)(gaddr & 0xFFFFFFFFull);               // global_addr[31:0]
  g0[3] = (unsigned)((gaddr >> 32) & 0x1FFFFFFull) | (2u << 30);  // addr[56:32] | type=2
  v8i g1;
  g1[0] = (int)((1u << 16)      // data_size = 2 bytes
              | (1u << 20)      // pad_enable
              | (4u << 22)      // pad_interval: every 32 DWORDs (=128B row)
              | (3u << 25));    // pad_amount: 4 DWORDs (=16B) -> row stride 144B = 72 bf16
  g1[1] = (int)(64u << 16);     // atomic_barrier_addr=0 | tensor_dim0[15:0]=64
  g1[2] = (int)(64u << 16);     // tensor_dim0[31:16]=0  | tensor_dim1[15:0]=64
  g1[3] = (int)(64u << 16);     // tensor_dim1[31:16]=0  | tile_dim0=64
  g1[4] = 64;                   // tile_dim1=64 | tile_dim2=0
  g1[5] = (int)stride_elems;    // tensor_dim0_stride[31:0]
  g1[6] = 0;                    // stride[47:32]=0 | tensor_dim1_stride lo=0
  g1[7] = 0;
  const v4i z4 = {0, 0, 0, 0};        // 2-D tensor: groups 2/3 unused
  const v8i z8 = {0, 0, 0, 0, 0, 0, 0, 0};
  __builtin_amdgcn_tensor_load_to_lds(g0, g1, z4, z4, z8, 0);
}

// ---------------- Kernel 1: qkv GEMM; Q->f32 [B,H,N,d], K->bf16 [B,H,N,d], V->bf16^T [B,H,d,N] ----
__global__ __launch_bounds__(256) void k_qkv(const float* __restrict__ x,
                                             const float* __restrict__ w,
                                             float* __restrict__ Qf,
                                             __bf16* __restrict__ Kbf,
                                             __bf16* __restrict__ Vt) {
  const int lane = threadIdx.x & 31, wave = threadIdx.x >> 5;
  const int half = lane >> 4, l15 = lane & 15;
  const int rowbase = blockIdx.x * 128 + wave * 16;   // 8192 rows total
  const int oc = blockIdx.y * 16 + l15;               // 768 output cols
  const float* arow = x + (size_t)(rowbase + l15) * 256;
  const float* brow = w + (size_t)oc * 256;
  v8f acc = {};
#pragma unroll
  for (int ks = 0; ks < 8; ++ks) {                    // K = 256 in steps of 32
    const int db = ks * 32 + half * 8;
    v16bf a, b;
#pragma unroll
    for (int t = 0; t < 8; ++t) {
      a[t] = (__bf16)arow[db + t];      a[t + 8] = (__bf16)arow[db + 16 + t];
      b[t] = (__bf16)brow[db + t];      b[t + 8] = (__bf16)brow[db + 16 + t];
    }
    acc = WMMA_BF16(a, b, acc);
  }
  const int which = oc >> 8;                 // 0:q 1:k 2:v
  const int h = (oc >> 6) & 3, d = oc & 63;
  const int grow0 = rowbase + half * 8;      // 8 consecutive rows, never crosses batch
  const size_t bh = (size_t)(grow0 >> 12) * 4 + h;
  const int n0 = grow0 & 4095;
  if (which == 0) {
#pragma unroll
    for (int r = 0; r < 8; ++r) Qf[(bh * 4096 + n0 + r) * 64 + d] = acc[r];
  } else if (which == 1) {
#pragma unroll
    for (int r = 0; r < 8; ++r) Kbf[(bh * 4096 + n0 + r) * 64 + d] = (__bf16)acc[r];
  } else {
    __bf16* dst = Vt + (bh * 64 + d) * 4096 + n0;     // transposed: contiguous in n
#pragma unroll
    for (int r = 0; r < 8; r += 2) {
      v2bf p; p[0] = (__bf16)acc[r]; p[1] = (__bf16)acc[r + 1];
      *(v2bf*)(dst + r) = p;
    }
  }
}

// ---------------- Kernel 2: combined per-channel scale Sc = invq * invk * temp ----------------
__global__ __launch_bounds__(128) void k_norm(const float* __restrict__ Qf,
                                              const __bf16* __restrict__ Kbf,
                                              const float* __restrict__ temp,
                                              float* __restrict__ Sc) {
  __shared__ float red[128];
  const int bh = blockIdx.x, t = threadIdx.x;
  const int d = t & 63;
  float s = 0.f;
  if (t < 64) {
    const float* src = Qf + ((size_t)bh * 4096) * 64 + d;
#pragma unroll 4
    for (int n = 0; n < 4096; ++n) { float v = src[(size_t)n * 64]; s = fmaf(v, v, s); }
  } else {
    const __bf16* src = Kbf + ((size_t)bh * 4096) * 64 + d;
#pragma unroll 4
    for (int n = 0; n < 4096; ++n) { float v = (float)src[(size_t)n * 64]; s = fmaf(v, v, s); }
  }
  red[t] = 1.f / fmaxf(sqrtf(s), 1e-12f);
  __syncthreads();
  if (t < 64) Sc[bh * 64 + t] = red[t] * red[t + 64] * temp[bh & 3];
}

// ---------------- Kernel 3: flash attention; TDM double-buffered K/V tiles -> LDS -------------
__global__ __launch_bounds__(256) void k_attn(const float* __restrict__ Qf,
                                              const __bf16* __restrict__ Kbf,
                                              const __bf16* __restrict__ Vt,
                                              const float* __restrict__ Sc,
                                              float* __restrict__ Y) {
  __shared__ __bf16 KS[2][64][72];   // [key m][channel d], row stride 72 via TDM pad
  __shared__ __bf16 VT[2][64][72];   // [channel dv][key m]
  const int bh = blockIdx.y;
  const int tid = threadIdx.x, wave = tid >> 5, lane = tid & 31;
  const int half = lane >> 4, l15 = lane & 15;
  const int n = blockIdx.x * 128 + wave * 16 + l15;    // this lane's query column
  const float* qrow = Qf + ((size_t)bh * 4096 + n) * 64;
  const float* scv = Sc + bh * 64;
  v16bf Bq0, Bq1;                                       // Q as B operand, full scale folded in
#pragma unroll
  for (int t = 0; t < 8; ++t) {
    const int d0 = half * 8 + t;
    Bq0[t]     = (__bf16)(qrow[d0]      * scv[d0]);
    Bq0[t + 8] = (__bf16)(qrow[d0 + 16] * scv[d0 + 16]);
    Bq1[t]     = (__bf16)(qrow[d0 + 32] * scv[d0 + 32]);
    Bq1[t + 8] = (__bf16)(qrow[d0 + 48] * scv[d0 + 48]);
  }
  v8f O0 = {}, O1 = {}, O2 = {}, O3 = {};
  const float NEG_INF = -__builtin_inff();
  float mprev = NEG_INF, lrun = 0.f;

  const unsigned long long kbase = (unsigned long long)(uintptr_t)(Kbf + ((size_t)bh * 4096) * 64);
  const unsigned long long vbase = (unsigned long long)(uintptr_t)(Vt + ((size_t)bh * 64) * 4096);
  const unsigned ldsK0 = (unsigned)(uintptr_t)&KS[0][0][0];
  const unsigned ldsK1 = (unsigned)(uintptr_t)&KS[1][0][0];
  const unsigned ldsV0 = (unsigned)(uintptr_t)&VT[0][0][0];
  const unsigned ldsV1 = (unsigned)(uintptr_t)&VT[1][0][0];

  if (wave == 0) {                      // prologue: tile 0 -> buffer 0
    tdm_load_2d(ldsK0, kbase, 64);
    tdm_load_2d(ldsV0, vbase, 4096);
  }

  for (int it = 0; it < 64; ++it) {
    const int buf = it & 1;
    if (wave == 0) __builtin_amdgcn_s_wait_tensorcnt(0);
    __syncthreads();                    // tile `it` resident for everyone
    if (wave == 0 && it + 1 < 64) {     // prefetch tile it+1 into the other buffer
      const unsigned long long ko = kbase + (unsigned long long)(it + 1) * (64ull * 64ull * 2ull);
      const unsigned long long vo = vbase + (unsigned long long)(it + 1) * (64ull * 2ull);
      tdm_load_2d(buf ? ldsK0 : ldsK1, ko, 64);
      tdm_load_2d(buf ? ldsV0 : ldsV1, vo, 4096);
    }

    v8f S0 = {}, S1 = {}, S2 = {}, S3 = {};             // S^T tile: 64 keys x 16 queries
#pragma unroll
    for (int ks = 0; ks < 2; ++ks) {
      const int db = ks * 32 + half * 8;
      const v16bf bq = ks ? Bq1 : Bq0;
      S0 = WMMA_BF16(lds2(&KS[buf][ 0 + l15][db], &KS[buf][ 0 + l15][db + 16]), bq, S0);
      S1 = WMMA_BF16(lds2(&KS[buf][16 + l15][db], &KS[buf][16 + l15][db + 16]), bq, S1);
      S2 = WMMA_BF16(lds2(&KS[buf][32 + l15][db], &KS[buf][32 + l15][db + 16]), bq, S2);
      S3 = WMMA_BF16(lds2(&KS[buf][48 + l15][db], &KS[buf][48 + l15][db + 16]), bq, S3);
    }
    // online softmax over keys: column n is per-lane -> 32 local values + one half-merge shuffle
    float mloc = NEG_INF;
#pragma unroll
    for (int r = 0; r < 8; ++r)
      mloc = fmaxf(mloc, fmaxf(fmaxf(S0[r], S1[r]), fmaxf(S2[r], S3[r])));
    mloc = fmaxf(mloc, __shfl_xor(mloc, 16, 32));
    const float mnew = fmaxf(mprev, mloc);
    const float aold = __expf(mprev - mnew);
    v16bf P0, P1;                                        // P directly in next-WMMA B layout
    float ls = 0.f;
#pragma unroll
    for (int r = 0; r < 8; ++r) {
      const float p0 = __expf(S0[r] - mnew), p1 = __expf(S1[r] - mnew);
      const float p2 = __expf(S2[r] - mnew), p3 = __expf(S3[r] - mnew);
      ls += (p0 + p1) + (p2 + p3);
      P0[r] = (__bf16)p0; P0[r + 8] = (__bf16)p1;
      P1[r] = (__bf16)p2; P1[r + 8] = (__bf16)p3;
    }
    ls += __shfl_xor(ls, 16, 32);
    lrun = lrun * aold + ls;
    mprev = mnew;
#pragma unroll
    for (int r = 0; r < 8; ++r) { O0[r] *= aold; O1[r] *= aold; O2[r] *= aold; O3[r] *= aold; }
#pragma unroll
    for (int ks = 0; ks < 2; ++ks) {                     // O^T += V^T . P
      const int mb = ks * 32 + half * 8;
      const v16bf bp = ks ? P1 : P0;
      O0 = WMMA_BF16(lds2(&VT[buf][ 0 + l15][mb], &VT[buf][ 0 + l15][mb + 16]), bp, O0);
      O1 = WMMA_BF16(lds2(&VT[buf][16 + l15][mb], &VT[buf][16 + l15][mb + 16]), bp, O1);
      O2 = WMMA_BF16(lds2(&VT[buf][32 + l15][mb], &VT[buf][32 + l15][mb + 16]), bp, O2);
      O3 = WMMA_BF16(lds2(&VT[buf][48 + l15][mb], &VT[buf][48 + l15][mb + 16]), bp, O3);
    }
    __syncthreads();                    // everyone done with `buf` before it is refilled
  }
  const float invl = 1.f / lrun;
  const int b_ = bh >> 2, h = bh & 3;
  // write in reference's permuted layout Y[b, d, h, n] so the final GEMM reads contiguously
#pragma unroll
  for (int r = 0; r < 8; ++r) {
    const int dv0 = half * 8 + r;
    Y[(((size_t)b_ * 64 + dv0     ) * 4 + h) * 4096 + n] = O0[r] * invl;
    Y[(((size_t)b_ * 64 + dv0 + 16) * 4 + h) * 4096 + n] = O1[r] * invl;
    Y[(((size_t)b_ * 64 + dv0 + 32) * 4 + h) * 4096 + n] = O2[r] * invl;
    Y[(((size_t)b_ * 64 + dv0 + 48) * 4 + h) * 4096 + n] = O3[r] * invl;
  }
}

// ---------------- Kernel 4: out = Y @ w_out^T + b_out (Y already in permuted flat layout) -------
__global__ __launch_bounds__(256) void k_proj(const float* __restrict__ Yl,
                                              const float* __restrict__ w,
                                              const float* __restrict__ bias,
                                              float* __restrict__ out) {
  const int lane = threadIdx.x & 31, wave = threadIdx.x >> 5;
  const int half = lane >> 4, l15 = lane & 15;
  const int rowbase = blockIdx.x * 128 + wave * 16;    // 8192 rows
  const int oc = blockIdx.y * 16 + l15;                // 256 cols
  const float* arow = Yl + (size_t)(rowbase + l15) * 256;
  const float* brow = w + (size_t)oc * 256;
  v8f acc = {};
#pragma unroll
  for (int ks = 0; ks < 8; ++ks) {
    const int db = ks * 32 + half * 8;
    v16bf a, b;
#pragma unroll
    for (int t = 0; t < 8; ++t) {
      a[t] = (__bf16)arow[db + t];  a[t + 8] = (__bf16)arow[db + 16 + t];
      b[t] = (__bf16)brow[db + t];  b[t + 8] = (__bf16)brow[db + 16 + t];
    }
    acc = WMMA_BF16(a, b, acc);
  }
  const float bo = bias[oc];
#pragma unroll
  for (int r = 0; r < 8; ++r) {
    const int i = rowbase + half * 8 + r;
    out[(size_t)i * 256 + oc] = acc[r] + bo;
  }
}

extern "C" void kernel_launch(void* const* d_in, const int* in_sizes, int n_in,
                              void* d_out, int out_size, void* d_ws, size_t ws_size,
                              hipStream_t stream) {
  const float* x    = (const float*)d_in[0];
  const float* wqkv = (const float*)d_in[1];
  const float* wout = (const float*)d_in[2];
  const float* bout = (const float*)d_in[3];
  const float* temp = (const float*)d_in[4];
  float* out = (float*)d_out;

  const size_t E = (size_t)2 * 4 * 4096 * 64;   // 2,097,152 elements per tensor
  char* ws = (char*)d_ws;
  float*  Qf  = (float*)ws;                 ws += E * sizeof(float);    // 8 MB
  __bf16* Kbf = (__bf16*)ws;                ws += E * sizeof(__bf16);   // 4 MB
  __bf16* Vt  = (__bf16*)ws;                ws += E * sizeof(__bf16);   // 4 MB
  float*  Sc  = (float*)ws;                 ws += 512 * sizeof(float);
  float*  Y   = (float*)ws;                                            // 8 MB

  k_qkv <<<dim3(64, 48), 256, 0, stream>>>(x, wqkv, Qf, Kbf, Vt);
  k_norm<<<dim3(8),      128, 0, stream>>>(Qf, Kbf, temp, Sc);
  k_attn<<<dim3(32, 8),  256, 0, stream>>>(Qf, Kbf, Vt, Sc, Y);
  k_proj<<<dim3(64, 16), 256, 0, stream>>>(Y, wout, bout, out);
}
